// NeuromorphicLM_88957362634982
// MI455X (gfx1250) — matrix operand order repaired and checked by reading, not compiled, not run on hardware
//
#include <hip/hip_runtime.h>
#include <hip/hip_bf16.h>
#include <math.h>

typedef __attribute__((ext_vector_type(16))) _Float16 v16h;
typedef __attribute__((ext_vector_type(8)))  _Float16 v8h;
typedef __attribute__((ext_vector_type(4)))  _Float16 v4h;
typedef __attribute__((ext_vector_type(8)))  float    v8f;
typedef __attribute__((ext_vector_type(4)))  float    v4f;

#define BSZ     4
#define NSEQ    256
#define VOCAB   32000
#define DMODEL  1024
#define BB      8
#define CCC     4
#define GG      32
#define DC      64
#define DM      64
#define BSB     32          // BS*Bb
#define NMEM    1024        // N*Cc
#define NCOL    32768       // BS*N*G  (column-space rows)
#define RSLOTS  128
#define MEMN    1024
#define CEM     16
#define NPASS   3

// ---------------- math helpers ----------------
__device__ __forceinline__ float sigm(float x)  { return 1.f / (1.f + __expf(-x)); }
__device__ __forceinline__ float softp(float x) { return fmaxf(x, 0.f) + log1pf(__expf(-fabsf(x))); }
__device__ __forceinline__ float geluf(float x) {
  return 0.5f * x * (1.f + tanhf(0.7978845608f * (x + 0.044715f * x * x * x)));
}

// ---------------- CDNA5 async VMEM->LDS staging (ASYNCcnt, ISA ch.8 / 15.18) ----------------
// Copies one 128-byte row (64 halfs) into LDS asynchronously; same IOFFSET applies to
// both the global source and the LDS destination per the VGLOBAL async encoding.
__device__ __forceinline__ void stage_row_async(_Float16* dst, const _Float16* src) {
  unsigned lds = (unsigned)(unsigned long long)dst;        // generic->LDS: addr[31:0]
  unsigned long long ga = (unsigned long long)src;
#pragma unroll
  for (int i = 0; i < 8; ++i)
    asm volatile("global_load_async_to_lds_b128 %0, %1, off offset:%2"
                 :: "v"(lds), "v"(ga), "i"(i * 16) : "memory");
}
__device__ __forceinline__ void wait_async0() {
  asm volatile("s_wait_asynccnt 0" ::: "memory");
}

// ---------------- WMMA tile loaders (CDNA5 ISA §7.12.2 layouts, wave32) ----------------
// A: 16x32 f16 tile, row-major source, contiguous 16B vector loads (global or LDS)
__device__ __forceinline__ v16h load_a16(const _Float16* A, int lda, int m0, int k0) {
  int lane = threadIdx.x & 31;
  const _Float16* p = A + (long)(m0 + (lane & 15)) * lda + k0 + ((lane & 16) ? 8 : 0);
  v8h lo = *(const v8h*)p;
  v8h hi = *(const v8h*)(p + 16);
  v16h a;
#pragma unroll
  for (int i = 0; i < 8; ++i) { a[i] = lo[i]; a[8 + i] = hi[i]; }
  return a;
}
// B: 32x16 tile, B stored transposed row-major [N x K] -> two 16B loads per lane at k0+8h and k0+16+8h
__device__ __forceinline__ v16h load_b16_nt(const _Float16* Bt, int ldb, int k0, int n0) {
  int lane = threadIdx.x & 31;
  const _Float16* p = Bt + (long)(n0 + (lane & 15)) * ldb + k0 + ((lane & 16) ? 8 : 0);
  v8h lo = *(const v8h*)p;
  v8h hi = *(const v8h*)(p + 16);
  v16h b;
#pragma unroll
  for (int i = 0; i < 8; ++i) { b[i] = lo[i]; b[8 + i] = hi[i]; }
  return b;
}
template<bool HAS_CIN, bool HAS_BIAS>
__device__ __forceinline__ void store_c16(float* C, int ldc, int m0, int n0, v8f acc,
                                          const float* Cin, const float* bias) {
  int lane = threadIdx.x & 31;
  int col  = n0 + (lane & 15);
  int rb   = m0 + ((lane & 16) ? 8 : 0);
#pragma unroll
  for (int j = 0; j < 8; ++j) {
    long idx = (long)(rb + j) * ldc + col;
    float v = acc[j];
    if (HAS_CIN)  v += Cin[idx];
    if (HAS_BIAS) v += bias[col];
    C[idx] = v;
  }
}
__device__ __forceinline__ v8f wmma_f16(v16h a, v16h b, v8f c) {
  return __builtin_amdgcn_wmma_f32_16x16x32_f16(false, a, false, b, (short)0, c, false, false);
}

// ---------------- NT GEMM: C[MxN] = A[MxK] * Bt[NxK]^T (+Cin)(+bias) ----------------
// grid = (M/16, N/(64*NTW)); 4 waves/block; each wave owns NTW adjacent 16x16 tiles.
template<int KC, int NTW, bool HAS_CIN, bool HAS_BIAS>
__global__ void __launch_bounds__(128)
gemm_nt(const _Float16* A, const _Float16* Bt, float* C, const float* Cin, const float* bias,
        int lda, int ldb, int ldc) {
  int wave = threadIdx.x >> 5;
  int m0 = blockIdx.x << 4;
  int n0 = (blockIdx.y * 4 + wave) * (16 * NTW);
  v8f z = {};
  v8f acc[NTW];
#pragma unroll
  for (int t = 0; t < NTW; ++t) acc[t] = z;
#pragma unroll 8
  for (int k0 = 0; k0 < KC; k0 += 32) {
    if (KC > 512 && (k0 & 255) == 0 && k0 + 512 < KC) {
      __builtin_prefetch((const void*)(Bt + (long)(n0 + (threadIdx.x & 15)) * ldb + k0 + 512), 0, 1);
      __builtin_prefetch((const void*)(A + (long)(m0 + (threadIdx.x & 15)) * lda + k0 + 512), 0, 1);
    }
    v16h a = load_a16(A, lda, m0, k0);
#pragma unroll
    for (int t = 0; t < NTW; ++t) {
      v16h b = load_b16_nt(Bt, ldb, k0, n0 + 16 * t);
      acc[t] = wmma_f16(a, b, acc[t]);
    }
  }
#pragma unroll
  for (int t = 0; t < NTW; ++t)
    store_c16<HAS_CIN, HAS_BIAS>(C, ldc, m0, n0 + 16 * t, acc[t], Cin, bias);
}

// ---------------- elementwise kernels ----------------
__global__ void cvt_f16_kernel(const float* s, _Float16* d, long n4) {   // n multiple of 4
  long i = (long)blockIdx.x * blockDim.x + threadIdx.x;
  if (i >= n4) return;
  v4f x = ((const v4f*)s)[i];
  v4h y;
#pragma unroll
  for (int j = 0; j < 4; ++j) y[j] = (_Float16)x[j];
  ((v4h*)d)[i] = y;
}
__global__ void gelu_cvt_kernel(const float* s, _Float16* d, long n4) {
  long i = (long)blockIdx.x * blockDim.x + threadIdx.x;
  if (i >= n4) return;
  v4f x = ((const v4f*)s)[i];
  v4h y;
#pragma unroll
  for (int j = 0; j < 4; ++j) y[j] = (_Float16)geluf(x[j]);
  ((v4h*)d)[i] = y;
}
__global__ void transpose_cvt_kernel(const float* W, _Float16* Wt, int K, int N) {
  long i = (long)blockIdx.x * blockDim.x + threadIdx.x;
  if (i >= (long)K * N) return;
  int n = (int)(i % N), k = (int)(i / N);
  Wt[(long)n * K + k] = (_Float16)W[i];
}
__global__ void combine_kernel(const float* h, const float* mlp, const float* rd, float* xout, long n) {
  long i = (long)blockIdx.x * blockDim.x + threadIdx.x;
  if (i < n) xout[i] = h[i] + mlp[i] + rd[i];
}
__global__ void blend_kernel(float* xcols, const float* xout, const float* lam_logit, int first, long n) {
  long i = (long)blockIdx.x * blockDim.x + threadIdx.x;
  if (i >= n) return;
  if (first) { xcols[i] = xout[i]; return; }
  float lam = sigm(lam_logit[0]);
  xcols[i] = (1.f - lam) * xcols[i] + lam * xout[i];
}
__global__ void novelty_kernel(const float* wn, const float* sp, const float* ms, float* nov, long n) {
  long i = (long)blockIdx.x * blockDim.x + threadIdx.x;
  if (i < n) nov[i] = wn[i] * sp[i] * fmaxf(1.f - ms[i], 0.f);
}
__global__ void embed_kernel(const int* ids, const float* emb, const float* pos, _Float16* xh) {
  long i = (long)blockIdx.x * blockDim.x + threadIdx.x;
  if (i >= (long)BSZ * NSEQ * DMODEL) return;
  int d = (int)(i & (DMODEL - 1));
  int t = (int)(i >> 10);
  int n = t & (NSEQ - 1);
  int tok = ids[t];
  xh[i] = (_Float16)(emb[(long)tok * DMODEL + d] + pos[(long)n * DMODEL + d]);
}
__global__ void init_state_kernel(const float* pmK0, const float* pmV0, const float* pma0,
                                  const float* emK0, const float* emV0, const float* emS0,
                                  const unsigned char* reset,
                                  float* pmK, float* pmV, float* pma,
                                  float* emK, float* emV, float* emS) {
  long i = (long)blockIdx.x * blockDim.x + threadIdx.x;
  if (i < (long)BSB * RSLOTS * DM) {
    int bm = (int)(i >> 13);
    int rz = reset[bm >> 3] ? 1 : 0;
    pmK[i] = pmK0[i];
    pmV[i] = rz ? 0.f : pmV0[i];
  }
  if (i < (long)BSB * RSLOTS) {
    int bm = (int)(i >> 7);
    pma[i] = reset[bm >> 3] ? 0.f : pma0[i];
  }
  if (i < (long)BSB * MEMN * DM) {
    emK[i] = emK0[i];
    emV[i] = emV0[i];
  }
  if (i < (long)BSB * MEMN) {
    int bm = (int)(i >> 10);
    emS[i] = reset[bm >> 3] ? 0.f : emS0[i];
  }
}

// ---------------- permutes (column-space <-> memory-space) ----------------
// mode 0: raw f32 -> dst ; mode 1: normalized -> dst & dsth ; mode 2: raw -> dst, normalized -> dsth
__global__ void to_mem_kernel(const float* src, float* dst, _Float16* dsth, int mode) {
  __shared__ float red[64];
  int row = blockIdx.x;               // bm*1024 + nm
  int d   = threadIdx.x;
  int bm = row >> 10, nm = row & 1023;
  int b = bm >> 3, bb = bm & 7, n = nm >> 2, cc = nm & 3;
  long rc = ((long)(b * NSEQ + n) * GG + bb * CCC + cc);
  float v = src[rc * 64 + d];
  float vn = v;
  if (mode != 0) {
    red[d] = v * v;
    __syncthreads();
    for (int s = 32; s > 0; s >>= 1) { if (d < s) red[d] += red[d + s]; __syncthreads(); }
    float nrm = sqrtf(red[0]) + 1e-6f;
    vn = v / nrm;
  }
  long o = (long)row * 64 + d;
  if (mode == 1) { dst[o] = vn; if (dsth) dsth[o] = (_Float16)vn; }
  else           { dst[o] = v;  if (mode == 2) dsth[o] = (_Float16)vn; }
}
__global__ void from_mem_kernel(const float* src, _Float16* dst) {
  int rc = blockIdx.x;                // column-space row
  int d  = threadIdx.x;
  int b = rc >> 13, rem = rc & 8191, n = rem >> 5, g = rem & 31;
  int bb = g >> 2, cc = g & 3;
  int bm = (b << 3) | bb, nm = (n << 2) | cc;
  dst[(long)rc * 64 + d] = (_Float16)src[((long)bm * NMEM + nm) * 64 + d];
}

// ---------------- scalar heads: gate / novelty-weight / surprise ----------------
__global__ void heads_kernel(const float* xout, const float* wg, const float* wn, const float* wsp,
                             float* gate_m, float* wn_m, float* sp_m) {
  __shared__ float sg[64], sn[64], ss[64];
  int rc = blockIdx.x, t = threadIdx.x;
  float x = xout[(long)rc * 64 + t];
  sg[t] = x * wg[t]; sn[t] = x * wn[t]; ss[t] = x * wsp[t];
  __syncthreads();
  for (int s = 32; s > 0; s >>= 1) {
    if (t < s) { sg[t] += sg[t + s]; sn[t] += sn[t + s]; ss[t] += ss[t + s]; }
    __syncthreads();
  }
  if (t == 0) {
    int b = rc >> 13, rem = rc & 8191, n = rem >> 5, g = rem & 31;
    int bb = g >> 2, cc = g & 3;
    long o = (long)((b << 3) | bb) * NMEM + ((n << 2) | cc);
    gate_m[o] = sigm(sg[0]);
    wn_m[o]   = sigm(sn[0]);
    sp_m[o]   = softp(ss[0]);
  }
}

// ---------------- PM attention read (128 slots, fused softmax) ----------------
__global__ void __launch_bounds__(128)
pm_attn_read_kernel(const float* qm, const float* pmK, const float* pmV, const float* pma, float* out) {
  __shared__ float q[64], sc[128], red[128];
  int bm = blockIdx.x >> 10, nm = blockIdx.x & 1023;
  int t = threadIdx.x;
  if (t < 64) q[t] = qm[((long)bm * NMEM + nm) * 64 + t];
  __syncthreads();
  const float* kr = pmK + ((long)bm * RSLOTS + t) * 64;
  float s = 0.f;
  for (int d = 0; d < 64; ++d) s += q[d] * kr[d];
  sc[t] = s; red[t] = s;
  __syncthreads();
  for (int st = 64; st > 0; st >>= 1) { if (t < st) red[t] = fmaxf(red[t], red[t + st]); __syncthreads(); }
  float mx = red[0];
  __syncthreads();
  float e = __expf(sc[t] - mx);
  sc[t] = e; red[t] = e;
  __syncthreads();
  for (int st = 64; st > 0; st >>= 1) { if (t < st) red[t] += red[t + st]; __syncthreads(); }
  float inv = 1.f / red[0];
  __syncthreads();
  sc[t] *= inv;
  __syncthreads();
  if (t < 64) {
    float acc = 0.f;
    for (int r = 0; r < RSLOTS; ++r)
      acc += sc[r] * pmV[((long)bm * RSLOTS + r) * 64 + t] * pma[(long)bm * RSLOTS + r];
    out[((long)bm * NMEM + nm) * 64 + t] = acc;
  }
}

// ---------------- EM prep: unit(em_K) f16 (NT) and (em_V*em_S)^T f16 ----------------
__global__ void em_prep_kernel(const float* emK, const float* emV, const float* emS,
                               _Float16* Knh, _Float16* Vst) {
  __shared__ float red[64];
  long row = blockIdx.x;              // bm*1024 + m
  int d = threadIdx.x;
  int bm = (int)(row >> 10), m = (int)(row & 1023);
  float k = emK[row * 64 + d];
  red[d] = k * k;
  __syncthreads();
  for (int s = 32; s > 0; s >>= 1) { if (d < s) red[d] += red[d + s]; __syncthreads(); }
  float nrm = sqrtf(red[0]) + 1e-6f;
  Knh[row * 64 + d] = (_Float16)(k / nrm);
  // transposed scaled values: Vst[bm][d][m]
  Vst[((long)bm * 64 + d) * MEMN + m] = (_Float16)(emV[row * 64 + d] * emS[row]);
}

// ---------------- EM attention: async-staged WMMA scores -> LDS softmax -> WMMA read ----------------
// Key panel (1024x64 f16 = 128KB) consumed in 8 chunks of 128 rows; chunk c+1 is DMA'd
// into LDS with GLOBAL_LOAD_ASYNC_TO_LDS_B128 while chunk c feeds the score WMMAs.
__global__ void __launch_bounds__(128)
em_attn_kernel(const _Float16* qmh, const _Float16* Knh, const _Float16* Vst, float* out) {
  __shared__ _Float16 S16[16 * 1024];          // 32 KB scores -> probabilities (f16)
  __shared__ _Float16 Kst[2][128 * 64];        // 2 x 16 KB staged key rows
  int bm = blockIdx.x >> 6, mt = blockIdx.x & 63;
  int m0 = mt << 4;
  const _Float16* A  = qmh + (long)bm * NMEM * 64;
  const _Float16* Kb = Knh + (long)bm * MEMN * 64;
  const _Float16* Vb = Vst + (long)bm * 64 * MEMN;
  int wave = threadIdx.x >> 5, lane = threadIdx.x & 31;
  int t = threadIdx.x;
  v16h a0 = load_a16(A, 64, m0, 0);
  v16h a1 = load_a16(A, 64, m0, 32);
  // prologue: stage chunk 0 (row t of 128)
  stage_row_async(&Kst[0][t * 64], Kb + (long)t * 64);
  for (int c = 0; c < 8; ++c) {
    wait_async0();
    __syncthreads();
    if (c + 1 < 8)
      stage_row_async(&Kst[(c + 1) & 1][t * 64], Kb + (long)((c + 1) * 128 + t) * 64);
    const _Float16* Kc = Kst[c & 1];
#pragma unroll
    for (int kt2 = 0; kt2 < 2; ++kt2) {
      int nloc = (wave * 2 + kt2) << 4;
      v8f acc = {};
      acc = wmma_f16(a0, load_b16_nt(Kc, 64, 0,  nloc), acc);
      acc = wmma_f16(a1, load_b16_nt(Kc, 64, 32, nloc), acc);
      int col = c * 128 + nloc + (lane & 15);
      int rb  = (lane & 16) ? 8 : 0;
#pragma unroll
      for (int j = 0; j < 8; ++j) S16[(rb + j) * 1024 + col] = (_Float16)(8.0f * acc[j]);
    }
  }
  __syncthreads();
  if (t < 16) {                       // per-row softmax, probs written in place (same thread)
    int r = t;
    float mx = -1e30f;
    for (int c2 = 0; c2 < 1024; ++c2) mx = fmaxf(mx, (float)S16[r * 1024 + c2]);
    float sum = 0.f;
    for (int c2 = 0; c2 < 1024; ++c2) sum += __expf((float)S16[r * 1024 + c2] - mx);
    float inv = 1.f / sum;
    for (int c2 = 0; c2 < 1024; ++c2)
      S16[r * 1024 + c2] = (_Float16)(__expf((float)S16[r * 1024 + c2] - mx) * inv);
  }
  __syncthreads();
  int n0 = wave << 4;                 // each wave: 16 output dims
  v8f acc = {};
#pragma unroll 8
  for (int k0 = 0; k0 < 1024; k0 += 32) {
    v16h pa = load_a16(S16, 1024, 0, k0);     // probabilities from LDS (ds_load_b128)
    v16h vb = load_b16_nt(Vb, MEMN, k0, n0);
    acc = wmma_f16(pa, vb, acc);
  }
  int col = n0 + (lane & 15);
  int rb  = (lane & 16) ? 8 : 0;
#pragma unroll
  for (int j = 0; j < 8; ++j)
    out[((long)bm * NMEM + m0 + rb + j) * 64 + col] = acc[j];
}

// ---------------- EM row-max (novelty): WMMA scores -> LDS -> row max ----------------
__global__ void __launch_bounds__(128)
em_rowmax_kernel(const _Float16* qh, const _Float16* Knh, float* maxsim) {
  __shared__ float S[16 * 1024];
  int bm = blockIdx.x >> 6, mt = blockIdx.x & 63;
  int m0 = mt << 4;
  const _Float16* A  = qh  + (long)bm * NMEM * 64;
  const _Float16* Kb = Knh + (long)bm * MEMN * 64;
  int wave = threadIdx.x >> 5, lane = threadIdx.x & 31;
  v16h a0 = load_a16(A, 64, m0, 0);
  v16h a1 = load_a16(A, 64, m0, 32);
  for (int kt = wave; kt < 64; kt += 4) {
    int n0 = kt << 4;
    v8f acc = {};
    acc = wmma_f16(a0, load_b16_nt(Kb, 64, 0,  n0), acc);
    acc = wmma_f16(a1, load_b16_nt(Kb, 64, 32, n0), acc);
    int col = n0 + (lane & 15);
    int rb  = (lane & 16) ? 8 : 0;
#pragma unroll
    for (int j = 0; j < 8; ++j) S[(rb + j) * 1024 + col] = acc[j];
  }
  __syncthreads();
  if (threadIdx.x < 16) {
    int r = threadIdx.x;
    float mx = -1e30f;
    for (int c = 0; c < 1024; ++c) mx = fmaxf(mx, S[r * 1024 + c]);
    maxsim[(long)bm * NMEM + m0 + r] = mx;
  }
}

// ---------------- PM routing: gr = gate * softmax(unit(k)·pm_K) ----------------
__global__ void __launch_bounds__(128)
route_kernel(const float* k_m, const float* pmK, const float* gate_m, float* gr) {
  __shared__ float kh[64], sc[128], red[128];
  int bm = blockIdx.x >> 10, nm = blockIdx.x & 1023;
  int t = threadIdx.x;
  if (t < 64) { float v = k_m[((long)bm * NMEM + nm) * 64 + t]; kh[t] = v; red[t] = v * v; }
  __syncthreads();
  if (t < 32) red[t] += red[t + 32];
  __syncthreads();
  if (t < 16) red[t] += red[t + 16];
  __syncthreads();
  if (t < 8) red[t] += red[t + 8];
  __syncthreads();
  if (t < 4) red[t] += red[t + 4];
  __syncthreads();
  if (t < 2) red[t] += red[t + 2];
  __syncthreads();
  if (t == 0) red[0] += red[1];
  __syncthreads();
  float inrm = 1.f / (sqrtf(red[0]) + 1e-6f);
  __syncthreads();
  const float* kr = pmK + ((long)bm * RSLOTS + t) * 64;
  float s = 0.f;
  for (int d = 0; d < 64; ++d) s += kh[d] * inrm * kr[d];
  sc[t] = s; red[t] = s;                      // TAU_ROUTE = 1.0
  __syncthreads();
  for (int st = 64; st > 0; st >>= 1) { if (t < st) red[t] = fmaxf(red[t], red[t + st]); __syncthreads(); }
  float mx = red[0];
  __syncthreads();
  float e = __expf(sc[t] - mx);
  sc[t] = e; red[t] = e;
  __syncthreads();
  for (int st = 64; st > 0; st >>= 1) { if (t < st) red[t] += red[t + st]; __syncthreads(); }
  float inv = 1.f / red[0];
  float gate = gate_m[(long)bm * NMEM + nm];
  gr[((long)bm * NMEM + nm) * RSLOTS + t] = gate * sc[t] * inv;
}

__global__ void elig_kernel(const float* gr, const float* k_m, const float* v_m,
                            float* eK, float* eV) {
  int bm = blockIdx.x >> 7, r = blockIdx.x & 127;
  int d = threadIdx.x;
  const float* grb = gr + (long)bm * NMEM * RSLOTS + r;
  const float* kb  = k_m + (long)bm * NMEM * 64 + d;
  const float* vb  = v_m + (long)bm * NMEM * 64 + d;
  float aK = 0.f, aV = 0.f;
  for (int nm = 0; nm < NMEM; ++nm) {
    float w = grb[(long)nm * RSLOTS];
    aK += w * kb[(long)nm * 64];
    aV += w * vb[(long)nm * 64];
  }
  long o = ((long)bm * RSLOTS + r) * 64 + d;
  eK[o] = aK; eV[o] = aV;
}

__global__ void pm_decay_kernel(float* pmV, float* pma) {
  long i = (long)blockIdx.x * blockDim.x + threadIdx.x;
  if (i < (long)BSB * RSLOTS * DM) pmV[i] *= 0.99f;
  if (i < (long)BSB * RSLOTS) pma[i] *= 0.99f;
}

__global__ void __launch_bounds__(128)
pmn_kernel(const float* eK, const float* pma, const float* W1, const float* b1,
           const float* W2, const float* b2, float* pm_g, float* slotw) {
  __shared__ float red[128], meand[64], hid[64], outsh[130];
  __shared__ float in0s, in1s;
  int bm = blockIdx.x, t = threadIdx.x;
  { const float* p = eK + ((long)bm * RSLOTS + t) * 64;
    float s = 0.f; for (int d = 0; d < 64; ++d) s += p[d] * p[d];
    red[t] = sqrtf(s); }
  __syncthreads();
  for (int s = 64; s > 0; s >>= 1) { if (t < s) red[t] += red[t + s]; __syncthreads(); }
  if (t == 0) in0s = red[0] / 128.f;
  __syncthreads();
  red[t] = pma[(long)bm * RSLOTS + t];
  __syncthreads();
  for (int s = 64; s > 0; s >>= 1) { if (t < s) red[t] += red[t + s]; __syncthreads(); }
  if (t == 0) in1s = red[0];
  if (t < 64) {
    float s = 0.f;
    for (int r = 0; r < RSLOTS; ++r) s += eK[((long)bm * RSLOTS + r) * 64 + t];
    meand[t] = s / 128.f;
  }
  __syncthreads();
  if (t < 64) {
    float a = b1[t] + in0s * W1[t] + in1s * W1[64 + t];
    for (int d = 0; d < 64; ++d) a += meand[d] * W1[(2 + d) * 64 + t];
    hid[t] = tanhf(a);
  }
  __syncthreads();
  for (int o = t; o < 130; o += 128) {
    float a = b2[o];
    for (int j = 0; j < 64; ++j) a += hid[j] * W2[j * 130 + o];
    outsh[o] = a;
  }
  __syncthreads();
  float g   = sigm(outsh[0]);
  float tau = softp(outsh[1]) + 0.5f;
  float sv  = outsh[2 + t] / tau;
  red[t] = sv;
  __syncthreads();
  for (int s = 64; s > 0; s >>= 1) { if (t < s) red[t] = fmaxf(red[t], red[t + s]); __syncthreads(); }
  float mx = red[0];
  __syncthreads();
  float e = __expf(sv - mx);
  red[t] = e;
  __syncthreads();
  for (int s = 64; s > 0; s >>= 1) { if (t < s) red[t] += red[t + s]; __syncthreads(); }
  slotw[(long)bm * RSLOTS + t] = e / red[0];
  if (t == 0) pm_g[bm] = g;
}

__global__ void pm_apply_kernel(float* pmK, float* pmV, float* pma,
                                const float* eK, const float* eV,
                                const float* pm_g, const float* slotw) {
  __shared__ float red[64];
  int bm = blockIdx.x >> 7, r = blockIdx.x & 127;
  int d = threadIdx.x;
  float upd = pm_g[bm] * slotw[(long)bm * RSLOTS + r];
  long o = ((long)bm * RSLOTS + r) * 64 + d;
  float k = pmK[o] + upd * eK[o];
  red[d] = k * k;
  __syncthreads();
  for (int s = 32; s > 0; s >>= 1) { if (d < s) red[d] += red[d + s]; __syncthreads(); }
  float nrm = sqrtf(red[0]) + 1e-6f;
  pmK[o] = k / nrm;
  pmV[o] += upd * eV[o];
  if (d == 0) pma[(long)bm * RSLOTS + r] += upd;
}

// ---------------- EM update: top-k novelty -> neuromodulated scatter write ----------------
__global__ void __launch_bounds__(128)
em_update_kernel(const float* novelty, const float* qn_m, const float* vn_m,
                 float* emK, float* emV, float* emS,
                 const float* W1, const float* b1, const float* W2, const float* b2) {
  __shared__ int   idx[16];
  __shared__ float csc[16], ckm[64], hid[64], outv[3], wstr[16];
  __shared__ int   slots[16];
  __shared__ float msc, ssum;
  int bm = blockIdx.x, t = threadIdx.x;
  __shared__ float red[128];
  const float* nv = novelty + (long)bm * NMEM;
  if (t == 0) {
    unsigned long long used[16];
    for (int i = 0; i < 16; ++i) used[i] = 0ull;
    float m = 0.f;
    for (int i = 0; i < CEM; ++i) {
      float best = -1e30f; int bi = 0;
      for (int c = 0; c < NMEM; ++c)
        if (!((used[c >> 6] >> (c & 63)) & 1ull)) { float v = nv[c]; if (v > best) { best = v; bi = c; } }
      used[bi >> 6] |= 1ull << (bi & 63);
      idx[i] = bi; csc[i] = best; m += best;
    }
    msc = m / CEM;
  }
  { float s = 0.f;
    for (int c = t; c < NMEM; c += 128) s += emS[(long)bm * NMEM + c];
    red[t] = s; }
  __syncthreads();
  for (int s = 64; s > 0; s >>= 1) { if (t < s) red[t] += red[t + s]; __syncthreads(); }
  if (t == 0) ssum = red[0];
  __syncthreads();
  if (t < 64) {
    float s = 0.f;
    for (int i = 0; i < CEM; ++i) s += qn_m[((long)bm * NMEM + idx[i]) * 64 + t];
    ckm[t] = s / CEM;
  }
  __syncthreads();
  if (t < 64) {
    float a = b1[t] + msc * W1[t] + ssum * W1[64 + t];
    for (int d = 0; d < 64; ++d) a += ckm[d] * W1[(2 + d) * 64 + t];
    hid[t] = tanhf(a);
  }
  __syncthreads();
  if (t < 3) {
    float a = b2[t];
    for (int j = 0; j < 64; ++j) a += hid[j] * W2[j * 3 + t];
    outv[t] = a;
  }
  __syncthreads();
  float g     = sigm(outv[0]);
  float tau   = softp(outv[1]) + 0.5f;
  float decay = 0.9f + 0.1f * sigm(outv[2]);
  for (int c = t; c < NMEM; c += 128) emS[(long)bm * NMEM + c] *= decay;
  __syncthreads();
  if (t < CEM) wstr[t] = g * sigm(csc[t] / tau);
  if (t == 0) {                          // weakest slots after decay
    unsigned long long used[16];
    for (int i = 0; i < 16; ++i) used[i] = 0ull;
    for (int i = 0; i < CEM; ++i) {
      float best = 1e30f; int bi = 0;
      for (int c = 0; c < NMEM; ++c)
        if (!((used[c >> 6] >> (c & 63)) & 1ull)) {
          float v = emS[(long)bm * NMEM + c];
          if (v < best) { best = v; bi = c; }
        }
      used[bi >> 6] |= 1ull << (bi & 63);
      slots[i] = bi;
    }
  }
  __syncthreads();
  if (t < CEM) {
    int sl = slots[t]; float w = wstr[t];
    const float* ck = qn_m + ((long)bm * NMEM + idx[t]) * 64;
    const float* cv = vn_m + ((long)bm * NMEM + idx[t]) * 64;
    float nrm = 0.f;
    for (int d = 0; d < 64; ++d) nrm += ck[d] * ck[d];
    nrm = sqrtf(nrm) + 1e-6f;
    float* Kr = emK + ((long)bm * MEMN + sl) * 64;
    float* Vr = emV + ((long)bm * MEMN + sl) * 64;
    for (int d = 0; d < 64; ++d) {
      Kr[d] = (1.f - w) * Kr[d] + w * (ck[d] / nrm);
      Vr[d] = (1.f - w) * Vr[d] + w * cv[d];
    }
    emS[(long)bm * NMEM + sl] += w;
  }
  __syncthreads();
  for (int c = t; c < NMEM; c += 128) emS[(long)bm * NMEM + c] *= 0.999f;
}

// ---------------- LayerNorm (writes f16 for the lm-head GEMM) ----------------
__global__ void __launch_bounds__(256)
layernorm_kernel(const float* x, const float* g, const float* b, _Float16* out) {
  __shared__ float red[256];
  __shared__ float mu_s, var_s;
  int row = blockIdx.x, t = threadIdx.x;
  float s = 0.f;
  for (int c = t; c < DMODEL; c += 256) s += x[(long)row * DMODEL + c];
  red[t] = s;
  __syncthreads();
  for (int st = 128; st > 0; st >>= 1) { if (t < st) red[t] += red[t + st]; __syncthreads(); }
  if (t == 0) mu_s = red[0] / DMODEL;
  __syncthreads();
  float mu = mu_s;
  float v = 0.f;
  for (int c = t; c < DMODEL; c += 256) { float d = x[(long)row * DMODEL + c] - mu; v += d * d; }
  red[t] = v;
  __syncthreads();
  for (int st = 128; st > 0; st >>= 1) { if (t < st) red[t] += red[t + st]; __syncthreads(); }
  if (t == 0) var_s = red[0] / DMODEL;
  __syncthreads();
  float inv = rsqrtf(var_s + 1e-5f);
  for (int c = t; c < DMODEL; c += 256)
    out[(long)row * DMODEL + c] =
        (_Float16)((x[(long)row * DMODEL + c] - mu) * inv * g[c] + b[c]);
}

// ==================== host launch ====================
static inline void* wsp_alloc(void* ws, size_t& off, size_t bytes) {
  void* p = (char*)ws + off;
  off += (bytes + 255) & ~(size_t)255;
  return p;
}

extern "C" void kernel_launch(void* const* d_in, const int* in_sizes, int n_in,
                              void* d_out, int out_size, void* d_ws, size_t ws_size,
                              hipStream_t stream) {
  const int*   ids   = (const int*)d_in[0];
  const unsigned char* reset = (const unsigned char*)d_in[1];
  const float* emb   = (const float*)d_in[2];
  const float* pos   = (const float*)d_in[3];
  const float* foW   = (const float*)d_in[4];
  const float* fob   = (const float*)d_in[5];
  const float* fiW   = (const float*)d_in[6];
  const float* fib   = (const float*)d_in[7];
  const float* lng   = (const float*)d_in[8];
  const float* lnb   = (const float*)d_in[9];
  const float* Wq    = (const float*)d_in[10];
  const float* Wk    = (const float*)d_in[11];
  const float* Wv    = (const float*)d_in[12];
  const float* Wqn   = (const float*)d_in[13];
  const float* Wvn   = (const float*)d_in[14];
  const float* wg    = (const float*)d_in[15];
  const float* wwn   = (const float*)d_in[16];
  const float* wsu   = (const float*)d_in[17];
  const float* WoPm  = (const float*)d_in[18];
  const float* WoEm  = (const float*)d_in[19];
  const float* mW1   = (const float*)d_in[20];
  const float* mW2   = (const float*)d_in[21];
  const float* pW1   = (const float*)d_in[22];
  const float* pb1   = (const float*)d_in[23];
  const float* pW2   = (const float*)d_in[24];
  const float* pb2   = (const float*)d_in[25];
  const float* eW1   = (const float*)d_in[26];
  const float* eb1   = (const float*)d_in[27];
  const float* eW2   = (const float*)d_in[28];
  const float* eb2   = (const float*)d_in[29];
  const float* lamb  = (const float*)d_in[30];
  const float* pmK0  = (const float*)d_in[31];
  const float* pmV0  = (const float*)d_in[32];
  const float* pma0  = (const float*)d_in[33];
  const float* emK0  = (const float*)d_in[34];
  const float* emV0  = (const float*)d_in[35];
  const float* emS0  = (const float*)d_in[36];

  const long EC = (long)NCOL * 64;    // 2,097,152 (== 1024 x 2048)

  size_t off = 0;
  _Float16* emb_h   = (_Float16*)wsp_alloc(d_ws, off, (size_t)VOCAB * DMODEL * 2);   // NT already
  _Float16* foWt    = (_Float16*)wsp_alloc(d_ws, off, (size_t)DMODEL * 2048 * 2);    // [2048 x 1024]
  _Float16* fiWt    = (_Float16*)wsp_alloc(d_ws, off, (size_t)2048 * DMODEL * 2);    // [1024 x 2048]
  _Float16* WqT     = (_Float16*)wsp_alloc(d_ws, off, 4096 * 2);
  _Float16* WkT     = (_Float16*)wsp_alloc(d_ws, off, 4096 * 2);
  _Float16* WvT     = (_Float16*)wsp_alloc(d_ws, off, 4096 * 2);
  _Float16* WqnT    = (_Float16*)wsp_alloc(d_ws, off, 4096 * 2);
  _Float16* WvnT    = (_Float16*)wsp_alloc(d_ws, off, 4096 * 2);
  _Float16* WoPmT   = (_Float16*)wsp_alloc(d_ws, off, 4096 * 2);
  _Float16* WoEmT   = (_Float16*)wsp_alloc(d_ws, off, 4096 * 2);
  _Float16* mW1T    = (_Float16*)wsp_alloc(d_ws, off, 8192 * 2);                      // [128 x 64]
  _Float16* mW2T    = (_Float16*)wsp_alloc(d_ws, off, 8192 * 2);                      // [64 x 128]
  _Float16* x_h     = (_Float16*)wsp_alloc(d_ws, off, (size_t)1024 * 1024 * 2);
  float*    xcols   = (float*)wsp_alloc(d_ws, off, EC * 4);
  float*    xout    = (float*)wsp_alloc(d_ws, off, EC * 4);
  _Float16* h_h     = (_Float16*)wsp_alloc(d_ws, off, EC * 2);
  _Float16* xout_h  = (_Float16*)wsp_alloc(d_ws, off, EC * 2);
  float*    coltmp  = (float*)wsp_alloc(d_ws, off, EC * 4);
  float*    xread   = (float*)wsp_alloc(d_ws, off, EC * 4);
  float*    mlp_hid = (float*)wsp_alloc(d_ws, off, EC * 2 * 4);
  _Float16* mlph_h  = (_Float16*)wsp_alloc(d_ws, off, EC * 2 * 2);
  float*    mlp_out = (float*)wsp_alloc(d_ws, off, EC * 4);
  float*    qm      = (float*)wsp_alloc(d_ws, off, EC * 4);
  _Float16* qm_h    = (_Float16*)wsp_alloc(d_ws, off, EC * 2);
  float*    k_m     = (float*)wsp_alloc(d_ws, off, EC * 4);
  float*    v_m     = (float*)wsp_alloc(d_ws, off, EC * 4);
  float*    qn_m    = (float*)wsp_alloc(d_ws, off, EC * 4);
  float*    vn_m    = (float*)wsp_alloc(d_ws, off, EC * 4);
  _Float16* qnhat_h = (_Float16*)wsp_alloc(d_ws, off, EC * 2);
  _Float16* rcol_h  = (_Float16*)wsp_alloc(d_ws, off, EC * 2);
  float*    gate_m  = (float*)wsp_alloc(d_ws, off, (size_t)BSB * NMEM * 4);
  float*    wn_m    = (float*)wsp_alloc(d_ws, off, (size_t)BSB * NMEM * 4);
  float*    sp_m    = (float*)wsp_alloc(d_ws, off, (size_t)BSB * NMEM * 4);
  float*    max_sim = (float*)wsp_alloc(d_ws, off, (size_t)BSB * NMEM * 4);
  float*    novlty  = (float*)wsp_alloc(d_ws, off, (size_t)BSB * NMEM * 4);
  float*    gr      = (float*)wsp_alloc(d_ws, off, (size_t)BSB * NMEM * RSLOTS * 4);
  float*    eK      = (float*)wsp_alloc(d_ws, off, (size_t)BSB * RSLOTS * DM * 4);
  float*    eV      = (float*)wsp_alloc(d_ws, off, (size_t)BSB * RSLOTS * DM * 4);
  float*    pm_g    = (float*)wsp_alloc(d_ws, off, (size_t)BSB * 4);
  float*    slotw   = (float*)wsp_alloc(d_ws, off, (size_t)BSB * RSLOTS * 4);
  float*    pmK     = (float*)wsp_alloc(d_ws, off, (size_t)BSB * RSLOTS * DM * 4);
  float*    pmV     = (float*)wsp_alloc(d_ws, off, (size_t)BSB * RSLOTS * DM * 4);
  float*    pma     = (float*)wsp_alloc(d_ws, off, (size_t)BSB * RSLOTS * 4);
  float*    emK     = (float*)wsp_alloc(d_ws, off, (size_t)BSB * MEMN * DM * 4);
  float*    emV     = (float*)wsp_alloc(d_ws, off, (size_t)BSB * MEMN * DM * 4);
  float*    emS     = (float*)wsp_alloc(d_ws, off, (size_t)BSB * MEMN * 4);
  _Float16* emKn_h  = (_Float16*)wsp_alloc(d_ws, off, (size_t)BSB * MEMN * DM * 2);
  _Float16* emVs_t  = (_Float16*)wsp_alloc(d_ws, off, (size_t)BSB * MEMN * DM * 2);  // transposed
  float*    pm_read = (float*)wsp_alloc(d_ws, off, EC * 4);
  float*    em_read = (float*)wsp_alloc(d_ws, off, EC * 4);
  float*    xfin    = (float*)wsp_alloc(d_ws, off, (size_t)1024 * DMODEL * 4);
  _Float16* xfin_h  = (_Float16*)wsp_alloc(d_ws, off, (size_t)1024 * DMODEL * 2);
  (void)ws_size; (void)n_in; (void)in_sizes; (void)out_size;

  auto cvt = [&](const float* s, _Float16* d, long n) {
    cvt_f16_kernel<<<(int)((n / 4 + 255) / 256), 256, 0, stream>>>(s, d, n / 4);
  };
  auto tcvt = [&](const float* W, _Float16* Wt, int K, int N) {
    transpose_cvt_kernel<<<(int)(((long)K * N + 255) / 256), 256, 0, stream>>>(W, Wt, K, N);
  };

  // ---- one-time conversions + state init ----
  cvt(emb, emb_h, (long)VOCAB * DMODEL);
  tcvt(foW, foWt, DMODEL, 2048);
  tcvt(fiW, fiWt, 2048, DMODEL);
  tcvt(Wq, WqT, 64, 64); tcvt(Wk, WkT, 64, 64); tcvt(Wv, WvT, 64, 64);
  tcvt(Wqn, WqnT, 64, 64); tcvt(Wvn, WvnT, 64, 64);
  tcvt(WoPm, WoPmT, 64, 64); tcvt(WoEm, WoEmT, 64, 64);
  tcvt(mW1, mW1T, 64, 128); tcvt(mW2, mW2T, 128, 64);
  init_state_kernel<<<(int)(((long)BSB * MEMN * DM + 255) / 256), 256, 0, stream>>>(
      pmK0, pmV0, pma0, emK0, emV0, emS0, reset, pmK, pmV, pma, emK, emV, emS);

  // ---- embed + fan-out ----
  embed_kernel<<<4096, 256, 0, stream>>>(ids, emb, pos, x_h);
  gemm_nt<1024, 2, false, true><<<dim3(64, 16), 128, 0, stream>>>(
      x_h, foWt, xcols, nullptr, fob, 1024, 1024, 2048);

  for (int pass = 0; pass < NPASS; ++pass) {
    // h = x_cols (column-space [32768,64]); f16 copy for GEMMs
    cvt(xcols, h_h, EC);

    // qm = unit(to_mem(h @ Wq))
    gemm_nt<64, 1, false, false><<<dim3(2048, 1), 128, 0, stream>>>(
        h_h, WqT, coltmp, nullptr, nullptr, 64, 64, 64);
    to_mem_kernel<<<BSB * NMEM, 64, 0, stream>>>(coltmp, qm, qm_h, 1);

    // PM read (fused softmax) and EM read (async-staged WMMA + LDS softmax)
    pm_attn_read_kernel<<<BSB * NMEM, 128, 0, stream>>>(qm, pmK, pmV, pma, pm_read);
    em_prep_kernel<<<BSB * MEMN, 64, 0, stream>>>(emK, emV, emS, emKn_h, emVs_t);
    em_attn_kernel<<<BSB * 64, 128, 0, stream>>>(qm_h, emKn_h, emVs_t, em_read);

    // x_read = from_mem(pm_read) @ Wo_pm + from_mem(em_read) @ Wo_em
    from_mem_kernel<<<NCOL, 64, 0, stream>>>(pm_read, rcol_h);
    gemm_nt<64, 1, false, false><<<dim3(2048, 1), 128, 0, stream>>>(
        rcol_h, WoPmT, xread, nullptr, nullptr, 64, 64, 64);
    from_mem_kernel<<<NCOL, 64, 0, stream>>>(em_read, rcol_h);
    gemm_nt<64, 1, true, false><<<dim3(2048, 1), 128, 0, stream>>>(
        rcol_h, WoEmT, xread, xread, nullptr, 64, 64, 64);

    // local MLP: gelu(h @ W1) @ W2
    gemm_nt<64, 2, false, false><<<dim3(2048, 1), 128, 0, stream>>>(
        h_h, mW1T, mlp_hid, nullptr, nullptr, 64, 64, 128);
    gelu_cvt_kernel<<<(int)((EC * 2 / 4 + 255) / 256), 256, 0, stream>>>(mlp_hid, mlph_h, EC * 2 / 4);
    gemm_nt<128, 1, false, false><<<dim3(2048, 1), 128, 0, stream>>>(
        mlph_h, mW2T, mlp_out, nullptr, nullptr, 128, 128, 64);

    // x_out = h + mlp + x_read
    combine_kernel<<<(int)((EC + 255) / 256), 256, 0, stream>>>(xcols, mlp_out, xread, xout, EC);
    cvt(xout, xout_h, EC);

    // eligibility / novelty heads from x_out
    gemm_nt<64, 1, false, false><<<dim3(2048, 1), 128, 0, stream>>>(
        xout_h, WkT, coltmp, nullptr, nullptr, 64, 64, 64);
    to_mem_kernel<<<BSB * NMEM, 64, 0, stream>>>(coltmp, k_m, nullptr, 0);
    gemm_nt<64, 1, false, false><<<dim3(2048, 1), 128, 0, stream>>>(
        xout_h, WvT, coltmp, nullptr, nullptr, 64, 64, 64);
    to_mem_kernel<<<BSB * NMEM, 64, 0, stream>>>(coltmp, v_m, nullptr, 0);
    gemm_nt<64, 1, false, false><<<dim3(2048, 1), 128, 0, stream>>>(
        xout_h, WqnT, coltmp, nullptr, nullptr, 64, 64, 64);
    to_mem_kernel<<<BSB * NMEM, 64, 0, stream>>>(coltmp, qn_m, qnhat_h, 2);
    gemm_nt<64, 1, false, false><<<dim3(2048, 1), 128, 0, stream>>>(
        xout_h, WvnT, coltmp, nullptr, nullptr, 64, 64, 64);
    to_mem_kernel<<<BSB * NMEM, 64, 0, stream>>>(coltmp, vn_m, nullptr, 0);
    heads_kernel<<<NCOL, 64, 0, stream>>>(xout, wg, wwn, wsu, gate_m, wn_m, sp_m);

    // PM: route -> decay -> eligibility -> neuromodulator -> apply
    route_kernel<<<BSB * NMEM, 128, 0, stream>>>(k_m, pmK, gate_m, gr);
    pm_decay_kernel<<<(int)(((long)BSB * RSLOTS * DM + 255) / 256), 256, 0, stream>>>(pmV, pma);
    elig_kernel<<<BSB * RSLOTS, 64, 0, stream>>>(gr, k_m, v_m, eK, eV);
    pmn_kernel<<<BSB, 128, 0, stream>>>(eK, pma, pW1, pb1, pW2, pb2, pm_g, slotw);
    pm_apply_kernel<<<BSB * RSLOTS, 64, 0, stream>>>(pmK, pmV, pma, eK, eV, pm_g, slotw);

    // EM: novelty (WMMA row-max) -> top-k scatter update
    em_rowmax_kernel<<<BSB * 64, 128, 0, stream>>>(qnhat_h, emKn_h, max_sim);
    novelty_kernel<<<(int)(((long)BSB * NMEM + 255) / 256), 256, 0, stream>>>(
        wn_m, sp_m, max_sim, novlty, (long)BSB * NMEM);
    em_update_kernel<<<BSB, 128, 0, stream>>>(
        novlty, qn_m, vn_m, emK, emV, emS, eW1, eb1, eW2, eb2);

    // residual blend
    blend_kernel<<<(int)((EC + 255) / 256), 256, 0, stream>>>(
        xcols, xout, lamb, pass == 0 ? 1 : 0, EC);
  }

  // ---- fan-in + layernorm + tied lm head ----
  cvt(xcols, h_h, EC);   // reuse as [1024 x 2048] f16
  gemm_nt<2048, 2, false, true><<<dim3(64, 8), 128, 0, stream>>>(
      h_h, fiWt, xfin, nullptr, fib, 2048, 2048, 1024);
  layernorm_kernel<<<1024, 256, 0, stream>>>(xfin, lng, lnb, xfin_h);
  gemm_nt<1024, 2, false, false><<<dim3(64, 250), 128, 0, stream>>>(
      xfin_h, emb_h, (float*)d_out, nullptr, nullptr, 1024, 1024, VOCAB);
}
